// Merge_Attention_43679817400667
// MI455X (gfx1250) — matrix operand order repaired and checked by reading, not compiled, not run on hardware
//
#include <hip/hip_runtime.h>

// ---------------------------------------------------------------------------
// Problem constants (reference: B=2, C=192, H=W=256, heads=4, hd=48)
// ---------------------------------------------------------------------------
#define B_    2
#define C_    192
#define N_    65536
#define HEADS 4
#define HD    48

// Tiling. All LDS strides are multiples of 8 ushorts (16 B) so fragment
// loads can use ds_load_b128 / global_load_b128.
#define TN    64          // columns per LDS tile
#define NITER 8           // n-tiles per workgroup in pass 1 (Gram regs persist)
#define SA    200         // LDS stride for [n][c] activation tiles (192 + 8)
#define SB    72          // LDS stride for [c][n] q/k tiles (64 + 8)
#define SV    224         // LDS stride for per-head padded v tiles (4*56)
#define SB2   584         // LDS stride for stacked B tile (576 + 8)

typedef __attribute__((ext_vector_type(16))) __bf16 bf16x16;
typedef __attribute__((ext_vector_type(2)))  __bf16 bf16x2;
typedef __attribute__((ext_vector_type(8)))  float  floatx8;

union AFrag {
  bf16x16 v;
  unsigned int u[8];
  uint4 q[2];
};

// Manual RNE float->bf16 (fallback when no native packed cvt is exposed)
__device__ __forceinline__ unsigned short f2bf_sw(float f) {
  union { float f; unsigned int u; } a; a.f = f;
  unsigned int u = a.u;
  unsigned int r = (u + 0x7FFFu + ((u >> 16) & 1u)) >> 16;   // RNE
  return (unsigned short)r;
}

// Single-value convert
__device__ __forceinline__ unsigned short f2bf(float f) {
#if __has_builtin(__builtin_amdgcn_cvt_pk_bf16_f32)
  union { bf16x2 v; unsigned short s[2]; } c;
  c.v = __builtin_amdgcn_cvt_pk_bf16_f32(f, 0.f);
  return c.s[0];
#else
  return f2bf_sw(f);
#endif
}

// Packed pair convert -> one dword (low = a, high = b)
__device__ __forceinline__ unsigned int pack2bf(float a, float b) {
#if __has_builtin(__builtin_amdgcn_cvt_pk_bf16_f32)
  union { bf16x2 v; unsigned int u; } c;
  c.v = __builtin_amdgcn_cvt_pk_bf16_f32(a, b);
  return c.u;
#else
  return (unsigned)f2bf_sw(a) | ((unsigned)f2bf_sw(b) << 16);
#endif
}

__device__ __forceinline__ floatx8 fzero() {
  floatx8 z;
#pragma unroll
  for (int r = 0; r < 8; ++r) z[r] = 0.f;
  return z;
}

// A fragment: 16x32 bf16, M x K. lane<16 -> row M=lane, K in {0..7,16..23};
// lane>=16 -> row M=lane-16, K in {8..15,24..31}; K pairs packed per dword.
// Two 16-byte contiguous chunks -> 2x b128 loads.
__device__ __forceinline__ void load_a(AFrag& f, const unsigned short* p, int ld, int lane) {
  const unsigned short* r = p + (lane & 15) * ld + (lane >> 4) * 8;
  f.q[0] = *(const uint4*)(r);
  f.q[1] = *(const uint4*)(r + 16);
}

// B fragment: 32x16 bf16, K x N, element (k, col) at p[col*ld + k] (K
// contiguous per column). lane<16 -> col=lane, K in {0..7,16..23}; lane>=16 ->
// col=lane-16, K in {8..15,24..31}. Two 16-byte chunks -> 2x b128 loads.
__device__ __forceinline__ void load_b(AFrag& f, const unsigned short* p, int ld, int lane) {
  const unsigned short* r = p + (lane & 15) * ld + (lane >> 4) * 8;
  f.q[0] = *(const uint4*)(r);
  f.q[1] = *(const uint4*)(r + 16);
}

__device__ __forceinline__ floatx8 wmma_bf16(const AFrag& a, const AFrag& b, floatx8 c) {
  return __builtin_amdgcn_wmma_f32_16x16x32_bf16(false, a.v, false, b.v, (short)0, c, false, false);
}

// ---------------------------------------------------------------------------
// Kernel 0: weight/bias fusion (tiny fp32 GEMMs, then bf16 convert)
// ---------------------------------------------------------------------------
__global__ __launch_bounds__(256) void k_prep_wq(const float* __restrict__ Wc,
                                                 const float* __restrict__ Wq,
                                                 unsigned short* __restrict__ Wqp) {
  int idx = blockIdx.x * 256 + threadIdx.x;
  if (idx >= C_ * C_) return;
  int o = idx / C_, c = idx % C_;
  float s = 0.f;
  for (int j = 0; j < C_; ++j) s += Wc[o * C_ + j] * Wq[j * C_ + c];
  Wqp[idx] = f2bf(s);
}

__global__ __launch_bounds__(256) void k_prep_wkv(const float* __restrict__ Wk,
                                                  const float* __restrict__ Wv,
                                                  unsigned short* __restrict__ Wkv) {
  int idx = blockIdx.x * 256 + threadIdx.x;
  if (idx >= 2 * C_ * C_) return;
  Wkv[idx] = f2bf(idx < C_ * C_ ? Wk[idx] : Wv[idx - C_ * C_]);
}

__global__ __launch_bounds__(256) void k_prep_wbig(const float* __restrict__ Wo,
                                                   const float* __restrict__ Wp1,
                                                   const float* __restrict__ Wp2,
                                                   unsigned short* __restrict__ Wbig) {
  int idx = blockIdx.x * 256 + threadIdx.x;
  if (idx >= C_ * 576) return;
  int o = idx / 576, k = idx % 576;
  float s;
  if (k < C_) {
    s = 0.f; for (int j = 0; j < C_; ++j) s += Wo[o * C_ + j] * Wp1[j * C_ + k];
  } else if (k < 2 * C_) {
    s = 0.f; for (int j = 0; j < C_; ++j) s += Wo[o * C_ + j] * Wp2[j * C_ + (k - C_)];
  } else {
    s = Wo[o * C_ + (k - 2 * C_)];
  }
  Wbig[idx] = f2bf(s);
}

__global__ __launch_bounds__(256) void k_prep_bias(const float* __restrict__ Wc,
                                                   const float* __restrict__ bq,
                                                   const float* __restrict__ bc,
                                                   const float* __restrict__ bk,
                                                   const float* __restrict__ bv,
                                                   const float* __restrict__ Wo,
                                                   const float* __restrict__ bp1,
                                                   const float* __restrict__ bp2,
                                                   const float* __restrict__ bo,
                                                   float* __restrict__ bqp,
                                                   float* __restrict__ bkv,
                                                   float* __restrict__ bop) {
  int i = blockIdx.x * 256 + threadIdx.x;
  if (i < C_) {
    float s = 0.f; for (int j = 0; j < C_; ++j) s += Wc[i * C_ + j] * bq[j];
    bqp[i] = 2.f * s + bc[i];
  } else if (i < 2 * C_) {
    bkv[i - C_] = bk[i - C_];
  } else if (i < 3 * C_) {
    bkv[i - C_] = bv[i - 2 * C_];
  } else if (i < 4 * C_) {
    int o = i - 3 * C_;
    float s = 0.f; for (int j = 0; j < C_; ++j) s += Wo[o * C_ + j] * (bp1[j] + bp2[j]);
    bop[o] = s + bo[o];
  }
}

// ---------------------------------------------------------------------------
// Kernel 1: q/k/v projections + sumsq + raw Gram accumulation
// grid (128, B), 256 threads, dynamic LDS = 162048 B
// ---------------------------------------------------------------------------
__global__ __launch_bounds__(256) void k_qkv(
    const float* __restrict__ x, const float* __restrict__ y,
    const unsigned short* __restrict__ Wqp, const unsigned short* __restrict__ Wkv,
    const float* __restrict__ bqp, const float* __restrict__ bkv,
    unsigned short* __restrict__ v1ws, unsigned short* __restrict__ v2ws,
    float* __restrict__ sumsq, float* __restrict__ gram1, float* __restrict__ gram2) {
  extern __shared__ unsigned short sm[];
  unsigned short* xt  = sm;                 // [TN][SA]  bf16(x)
  unsigned short* yt  = xt  + TN * SA;      // [TN][SA]  bf16(y)
  unsigned short* st  = yt  + TN * SA;      // [TN][SA]  bf16(x+y)
  unsigned short* qt  = st  + TN * SA;      // [C_][SB]  q tile
  unsigned short* k1t = qt  + C_ * SB;      // [C_][SB]  k1 tile
  unsigned short* k2t = k1t + C_ * SB;      // [C_][SB]  k2 tile
  float* rowsq = (float*)(k2t + C_ * SB);   // [576] : q(0..191) k1(192..) k2(384..)

  const int tid = threadIdx.x, lane = tid & 31, half = lane >> 4;
  // wave index is uniform across the wave: force it scalar so the tile loops
  // and ri<12 branch are SALU/s_cbranch, not exec-mask games.
  const int wave = __builtin_amdgcn_readfirstlane(tid) >> 5;
  const int b = blockIdx.y;

  for (int i = tid; i < 576; i += 256) rowsq[i] = 0.f;

  floatx8 gacc[9];
#pragma unroll
  for (int j = 0; j < 9; ++j) gacc[j] = fzero();

  for (int it = 0; it < NITER; ++it) {
    const int n0 = (blockIdx.x * NITER + it) * TN;
    __syncthreads();
    // load x, y (float4 -> global_load_b128); build bf16 tiles in [n][c]
    // layout (B-operand friendly). Prefetch next n-tile per 128B line.
    for (int idx = tid; idx < C_ * (TN / 4); idx += 256) {
      int c = idx / (TN / 4), n = (idx % (TN / 4)) * 4;
      const size_t gi = ((size_t)b * C_ + c) * N_ + n0 + n;
      const float4 xv = *(const float4*)&x[gi];
      const float4 yv = *(const float4*)&y[gi];
      if ((n & 31) == 0 && it + 1 < NITER) {
        __builtin_prefetch(&x[gi + TN], 0, 0);
        __builtin_prefetch(&y[gi + TN], 0, 0);
      }
      xt[(n + 0) * SA + c] = f2bf(xv.x);
      xt[(n + 1) * SA + c] = f2bf(xv.y);
      xt[(n + 2) * SA + c] = f2bf(xv.z);
      xt[(n + 3) * SA + c] = f2bf(xv.w);
      yt[(n + 0) * SA + c] = f2bf(yv.x);
      yt[(n + 1) * SA + c] = f2bf(yv.y);
      yt[(n + 2) * SA + c] = f2bf(yv.z);
      yt[(n + 3) * SA + c] = f2bf(yv.w);
      st[(n + 0) * SA + c] = f2bf(xv.x + yv.x);
      st[(n + 1) * SA + c] = f2bf(xv.y + yv.y);
      st[(n + 2) * SA + c] = f2bf(xv.z + yv.z);
      st[(n + 3) * SA + c] = f2bf(xv.w + yv.w);
    }
    __syncthreads();

    // stacked [Wk;Wv] GEMMs: g=0 on x (-> k1,v1), g=1 on y (-> k2,v2)
    for (int g = 0; g < 2; ++g) {
      const unsigned short* bt = g ? yt : xt;
      unsigned short* kt = g ? k2t : k1t;
      unsigned short* vws = g ? v2ws : v1ws;
      const int sqoff = g ? 384 : 192;
      for (int t = wave; t < 96; t += 8) {      // 24 row-tiles x 4 col-tiles
        const int ri = t >> 2, ci = t & 3;
        floatx8 acc = fzero();
#pragma unroll
        for (int kk = 0; kk < 6; ++kk) {
          AFrag a, bb;
          load_a(a, Wkv + (ri * 16) * C_ + kk * 32, C_, lane);
          load_b(bb, bt + (ci * 16) * SA + kk * 32, SA, lane);
          acc = wmma_bf16(a, bb, acc);
        }
        const int row0 = ri * 16 + half * 8;
        const int col = ci * 16 + (lane & 15);
        if (ri < 12) {                           // k rows 0..191
#pragma unroll
          for (int r = 0; r < 8; ++r) {
            float vv = acc[r] + bkv[row0 + r];
            atomicAdd(&rowsq[sqoff + row0 + r], vv * vv);
            kt[(row0 + r) * SB + col] = f2bf(vv);
          }
        } else {                                 // v rows 192..383 -> global ws
          const int o = row0 - C_;
          uint4 w;
          w.x = pack2bf(acc[0] + bkv[row0 + 0], acc[1] + bkv[row0 + 1]);
          w.y = pack2bf(acc[2] + bkv[row0 + 2], acc[3] + bkv[row0 + 3]);
          w.z = pack2bf(acc[4] + bkv[row0 + 4], acc[5] + bkv[row0 + 5]);
          w.w = pack2bf(acc[6] + bkv[row0 + 6], acc[7] + bkv[row0 + 7]);
          *(uint4*)(vws + ((size_t)b * N_ + n0 + col) * C_ + o) = w;
        }
      }
    }

    // q = (Wc·Wq)(x+y) + bq'
    for (int t = wave; t < 48; t += 8) {
      const int ri = t >> 2, ci = t & 3;
      floatx8 acc = fzero();
#pragma unroll
      for (int kk = 0; kk < 6; ++kk) {
        AFrag a, bb;
        load_a(a, Wqp + (ri * 16) * C_ + kk * 32, C_, lane);
        load_b(bb, st + (ci * 16) * SA + kk * 32, SA, lane);
        acc = wmma_bf16(a, bb, acc);
      }
      const int row0 = ri * 16 + half * 8;
      const int col = ci * 16 + (lane & 15);
#pragma unroll
      for (int r = 0; r < 8; ++r) {
        float vv = acc[r] + bqp[row0 + r];
        atomicAdd(&rowsq[row0 + r], vv * vv);
        qt[(row0 + r) * SB + col] = f2bf(vv);
      }
    }
    __syncthreads();

    // Gram accumulation (per head, q·k^T over this n-tile), regs persist
#pragma unroll
    for (int j = 0; j < 9; ++j) {
      const int f = wave + 8 * j;                // 0..71
      const int g2 = (f >= 36);
      const int rem = f - 36 * g2;
      const int h = rem / 9, p = rem % 9;
      const int ri = p / 3, ci = p % 3;
      const unsigned short* kt = g2 ? k2t : k1t;
#pragma unroll
      for (int kk = 0; kk < 2; ++kk) {
        AFrag a, bb;
        load_a(a, qt + (h * HD + ri * 16) * SB + kk * 32, SB, lane);
        load_b(bb, kt + (h * HD + ci * 16) * SB + kk * 32, SB, lane);
        gacc[j] = wmma_bf16(a, bb, gacc[j]);
      }
    }
  }
  __syncthreads();

  for (int i = tid; i < 576; i += 256) atomicAdd(&sumsq[b * 576 + i], rowsq[i]);

#pragma unroll
  for (int j = 0; j < 9; ++j) {
    const int f = wave + 8 * j;
    const int g2 = (f >= 36);
    const int rem = f - 36 * g2;
    const int h = rem / 9, p = rem % 9;
    const int ri = p / 3, ci = p % 3;
    float* G = (g2 ? gram2 : gram1) + (size_t)(b * HEADS + h) * HD * HD;
#pragma unroll
    for (int r = 0; r < 8; ++r)
      atomicAdd(&G[(ri * 16 + half * 8 + r) * HD + ci * 16 + (lane & 15)], gacc[j][r]);
  }
}

// ---------------------------------------------------------------------------
// Kernel 2: normalize Gram by L2 norms (eps clamp), x temperature, softmax,
// store zero-padded 48x64 bf16 attention matrices. grid 16, 64 threads.
// ---------------------------------------------------------------------------
__global__ __launch_bounds__(64) void k_softmax(
    const float* __restrict__ gram1, const float* __restrict__ gram2,
    const float* __restrict__ sumsq, const float* __restrict__ temp,
    unsigned short* __restrict__ A1bf, unsigned short* __restrict__ A2bf) {
  const int id = blockIdx.x;
  const int g = id & 1, h = (id >> 1) & 3, b = id >> 3;
  const float* G = (g ? gram2 : gram1) + (size_t)(b * HEADS + h) * HD * HD;
  unsigned short* A = (g ? A2bf : A1bf) + (size_t)(b * HEADS + h) * HD * 64;
  const float t = temp[h];
  const float* sq = sumsq + b * 576;
  const int c = threadIdx.x;
  if (c < HD) {
    float nq = fmaxf(sqrtf(sq[h * HD + c]), 1e-12f);
    const int ko = (g ? 384 : 192) + h * HD;
    float l[HD];
    float mx = -3.4e38f;
    for (int d = 0; d < HD; ++d) {
      float nk = fmaxf(sqrtf(sq[ko + d]), 1e-12f);
      float vv = t * G[c * HD + d] / (nq * nk);
      l[d] = vv;
      mx = fmaxf(mx, vv);
    }
    float s = 0.f;
    for (int d = 0; d < HD; ++d) { float e = expf(l[d] - mx); l[d] = e; s += e; }
    float inv = 1.f / s;
    for (int d = 0; d < HD; ++d) A[c * 64 + d] = f2bf(l[d] * inv);
    for (int d = HD; d < 64; ++d) A[c * 64 + d] = 0;
  }
}

// ---------------------------------------------------------------------------
// Kernel 3: u = attn @ v per head, then out = [WoWp1|WoWp2|Wo]·[u1;u2;x+y]+b'
// grid (1024, B), 256 threads, dynamic LDS = 132096 B
// ---------------------------------------------------------------------------
__global__ __launch_bounds__(256) void k_attn_out(
    const float* __restrict__ x, const float* __restrict__ y,
    const unsigned short* __restrict__ v1ws, const unsigned short* __restrict__ v2ws,
    const unsigned short* __restrict__ A1bf, const unsigned short* __restrict__ A2bf,
    const unsigned short* __restrict__ Wbig, const float* __restrict__ bop,
    float* __restrict__ out) {
  extern __shared__ unsigned short sm[];
  unsigned short* vt1 = sm;                 // [TN][SV] per-head padded v1
  unsigned short* vt2 = vt1 + TN * SV;      // [TN][SV]
  unsigned short* bt  = vt2 + TN * SV;      // [TN][SB2] stacked B: u1|u2|x+y

  const int tid = threadIdx.x, lane = tid & 31, half = lane >> 4;
  const int wave = __builtin_amdgcn_readfirstlane(tid) >> 5;
  const int b = blockIdx.y;
  const int n0 = blockIdx.x * TN;

  // v tiles: ws is [b][n][192] bf16; scatter into per-head padded [n][4][56]
  for (int idx = tid; idx < TN * (C_ / 2); idx += 256) {
    int n = idx / (C_ / 2), cw = idx % (C_ / 2);
    int c = cw * 2;
    int h = c / HD, d = c - h * HD;
    unsigned int u1 = *(const unsigned int*)(v1ws + ((size_t)b * N_ + n0 + n) * C_ + c);
    unsigned int u2 = *(const unsigned int*)(v2ws + ((size_t)b * N_ + n0 + n) * C_ + c);
    *(unsigned int*)(vt1 + n * SV + h * 56 + d) = u1;
    *(unsigned int*)(vt2 + n * SV + h * 56 + d) = u2;
  }
  // zero the K-padding (d = 48..55 per head) so K=64 WMMA is exact
  for (int idx = tid; idx < TN * 16; idx += 256) {
    int n = idx / 16, j = idx % 16;
    int h = j >> 2, q = j & 3;
    *(unsigned int*)(vt1 + n * SV + h * 56 + 48 + q * 2) = 0;
    *(unsigned int*)(vt2 + n * SV + h * 56 + 48 + q * 2) = 0;
  }
  // residual stream (x+y) -> bt columns 384..575 (float4 global loads)
  for (int idx = tid; idx < C_ * (TN / 4); idx += 256) {
    int c = idx / (TN / 4), n = (idx % (TN / 4)) * 4;
    const size_t gi = ((size_t)b * C_ + c) * N_ + n0 + n;
    const float4 xv = *(const float4*)&x[gi];
    const float4 yv = *(const float4*)&y[gi];
    bt[(n + 0) * SB2 + 384 + c] = f2bf(xv.x + yv.x);
    bt[(n + 1) * SB2 + 384 + c] = f2bf(xv.y + yv.y);
    bt[(n + 2) * SB2 + 384 + c] = f2bf(xv.z + yv.z);
    bt[(n + 3) * SB2 + 384 + c] = f2bf(xv.w + yv.w);
  }
  __syncthreads();

  // u = attn @ v : 96 fragments (u1: 48, u2: 48); K = 48 (+16 zero pad)
  for (int t = wave; t < 96; t += 8) {
    const int g = (t >= 48);
    const int tt = t - 48 * g;
    const int h = tt / 12, p = tt % 12;
    const int ri = p >> 2, ci = p & 3;
    const unsigned short* A = (g ? A2bf : A1bf) + (size_t)(b * HEADS + h) * HD * 64;
    const unsigned short* vt = g ? vt2 : vt1;
    floatx8 acc = fzero();
#pragma unroll
    for (int kk = 0; kk < 2; ++kk) {
      AFrag a, bb;
      load_a(a, A + (ri * 16) * 64 + kk * 32, 64, lane);
      load_b(bb, vt + (ci * 16) * SV + h * 56 + kk * 32, SV, lane);
      acc = wmma_bf16(a, bb, acc);
    }
    const int n = ci * 16 + (lane & 15);
    const int c0 = g * C_ + h * HD + ri * 16 + half * 8;
    uint4 w;
    w.x = pack2bf(acc[0], acc[1]);
    w.y = pack2bf(acc[2], acc[3]);
    w.z = pack2bf(acc[4], acc[5]);
    w.w = pack2bf(acc[6], acc[7]);
    *(uint4*)(bt + n * SB2 + c0) = w;
  }
  __syncthreads();

  // out = Wbig (192x576) x bt (576xTN) + bop
  for (int t = wave; t < 48; t += 8) {
    const int ri = t >> 2, ci = t & 3;
    floatx8 acc = fzero();
#pragma unroll 6
    for (int kk = 0; kk < 18; ++kk) {
      AFrag a, bb;
      load_a(a, Wbig + (ri * 16) * 576 + kk * 32, 576, lane);
      load_b(bb, bt + (ci * 16) * SB2 + kk * 32, SB2, lane);
      acc = wmma_bf16(a, bb, acc);
    }
    const int row0 = ri * 16 + half * 8;
    const int col = n0 + ci * 16 + (lane & 15);
#pragma unroll
    for (int r = 0; r < 8; ++r)
      out[((size_t)b * C_ + row0 + r) * N_ + col] = acc[r] + bop[row0 + r];
  }
}

// ---------------------------------------------------------------------------
// Host-side launch
// ---------------------------------------------------------------------------
extern "C" void kernel_launch(void* const* d_in, const int* in_sizes, int n_in,
                              void* d_out, int out_size, void* d_ws, size_t ws_size,
                              hipStream_t stream) {
  (void)in_sizes; (void)n_in; (void)out_size;
  const float* x   = (const float*)d_in[0];
  const float* y   = (const float*)d_in[1];
  const float* Wq  = (const float*)d_in[2];
  const float* bq  = (const float*)d_in[3];
  const float* Wk  = (const float*)d_in[4];
  const float* bk  = (const float*)d_in[5];
  const float* Wv  = (const float*)d_in[6];
  const float* bv  = (const float*)d_in[7];
  const float* Wc  = (const float*)d_in[8];
  const float* bc  = (const float*)d_in[9];
  const float* Wp1 = (const float*)d_in[10];
  const float* bp1 = (const float*)d_in[11];
  const float* Wp2 = (const float*)d_in[12];
  const float* bp2 = (const float*)d_in[13];
  const float* Wo  = (const float*)d_in[14];
  const float* bo  = (const float*)d_in[15];
  const float* temp = (const float*)d_in[16];
  float* out = (float*)d_out;

  // workspace layout (bytes); every block 16B-aligned
  const size_t SZ_V    = (size_t)B_ * N_ * C_ * 2;     // bf16 v tensor
  const size_t OFF_V1  = 0;
  const size_t OFF_V2  = OFF_V1 + SZ_V;
  const size_t OFF_WQP = OFF_V2 + SZ_V;                            // 192x192 bf16
  const size_t OFF_WKV = OFF_WQP + (size_t)C_ * C_ * 2;            // 384x192 bf16
  const size_t OFF_WBG = OFF_WKV + (size_t)2 * C_ * C_ * 2;        // 192x576 bf16
  const size_t OFF_BQP = OFF_WBG + (size_t)C_ * 576 * 2;           // 192 f32
  const size_t OFF_BKV = OFF_BQP + (size_t)C_ * 4;                 // 384 f32
  const size_t OFF_BOP = OFF_BKV + (size_t)2 * C_ * 4;             // 192 f32
  const size_t OFF_SSQ = OFF_BOP + (size_t)C_ * 4;                 // 2x576 f32
  const size_t OFF_G1  = OFF_SSQ + (size_t)B_ * 576 * 4;           // 2x4x48x48 f32
  const size_t OFF_G2  = OFF_G1 + (size_t)B_ * HEADS * HD * HD * 4;
  const size_t OFF_A1  = OFF_G2 + (size_t)B_ * HEADS * HD * HD * 4; // 2x4x48x64 bf16
  const size_t OFF_A2  = OFF_A1 + (size_t)B_ * HEADS * HD * 64 * 2;
  const size_t TOTAL   = OFF_A2 + (size_t)B_ * HEADS * HD * 64 * 2;
  if (ws_size < TOTAL) return;

  char* w = (char*)d_ws;
  unsigned short* v1ws = (unsigned short*)(w + OFF_V1);
  unsigned short* v2ws = (unsigned short*)(w + OFF_V2);
  unsigned short* Wqp  = (unsigned short*)(w + OFF_WQP);
  unsigned short* Wkv  = (unsigned short*)(w + OFF_WKV);
  unsigned short* Wbig = (unsigned short*)(w + OFF_WBG);
  float* bqp   = (float*)(w + OFF_BQP);
  float* bkv   = (float*)(w + OFF_BKV);
  float* bop   = (float*)(w + OFF_BOP);
  float* sumsq = (float*)(w + OFF_SSQ);
  float* gram1 = (float*)(w + OFF_G1);
  float* gram2 = (float*)(w + OFF_G2);
  unsigned short* A1bf = (unsigned short*)(w + OFF_A1);
  unsigned short* A2bf = (unsigned short*)(w + OFF_A2);

  // zero the accumulators (sumsq + gram1 + gram2 are contiguous)
  hipMemsetAsync(w + OFF_SSQ, 0, OFF_A1 - OFF_SSQ, stream);

  k_prep_wq  <<<(C_ * C_ + 255) / 256, 256, 0, stream>>>(Wc, Wq, Wqp);
  k_prep_wkv <<<(2 * C_ * C_ + 255) / 256, 256, 0, stream>>>(Wk, Wv, Wkv);
  k_prep_wbig<<<(C_ * 576 + 255) / 256, 256, 0, stream>>>(Wo, Wp1, Wp2, Wbig);
  k_prep_bias<<<(4 * C_ + 255) / 256, 256, 0, stream>>>(Wc, bq, bc, bk, bv, Wo, bp1, bp2, bo,
                                                        bqp, bkv, bop);

  const unsigned K1_LDS = (3u * TN * SA + 3u * C_ * SB) * 2u + 576u * 4u; // 162048
  dim3 g1(N_ / (TN * NITER), B_);
  k_qkv<<<g1, 256, K1_LDS, stream>>>(x, y, Wqp, Wkv, bqp, bkv, v1ws, v2ws,
                                     sumsq, gram1, gram2);

  k_softmax<<<B_ * HEADS * 2, 64, 0, stream>>>(gram1, gram2, sumsq, temp, A1bf, A2bf);

  const unsigned K3_LDS = (2u * TN * SV + (unsigned)TN * SB2) * 2u;       // 132096
  dim3 g3(N_ / TN, B_);
  k_attn_out<<<g3, 256, K3_LDS, stream>>>(x, y, v1ws, v2ws, A1bf, A2bf, Wbig, bop, out);
}
